// GCNEncoder_48533130445492
// MI455X (gfx1250) — compile-verified, matrix operand order checked by reading, not run on hardware
//
#include <hip/hip_runtime.h>

#define NFEAT 256
#define NHID  256
#define KPAD  264   // LDS column stride in bf16 (256 + 8 pad -> bank-conflict-free)

typedef __attribute__((ext_vector_type(16))) __bf16 v16bf;
typedef __attribute__((ext_vector_type(8)))  float  v8f;

union BFrag { uint4 u[2]; v16bf v; };
struct AFrag { float4 f[4]; };

__device__ __forceinline__ __bf16 f2bf(float x) { return (__bf16)x; }

__device__ __forceinline__ AFrag load_a(const float* xp) {
  AFrag r;
  r.f[0] = ((const float4*)xp)[0];
  r.f[1] = ((const float4*)xp)[1];
  r.f[2] = ((const float4*)(xp + 16))[0];
  r.f[3] = ((const float4*)(xp + 16))[1];
  return r;
}

__device__ __forceinline__ BFrag load_b_lds(const __bf16* wp) {
  BFrag r;
  r.u[0] = *(const uint4*)wp;          // ds_load_b128
  r.u[1] = *(const uint4*)(wp + 16);   // ds_load_b128 (+32 B)
  return r;
}

__device__ __forceinline__ v16bf cvt_a(const AFrag& q) {
  v16bf a;
  a[0]  = f2bf(q.f[0].x); a[1]  = f2bf(q.f[0].y); a[2]  = f2bf(q.f[0].z); a[3]  = f2bf(q.f[0].w);
  a[4]  = f2bf(q.f[1].x); a[5]  = f2bf(q.f[1].y); a[6]  = f2bf(q.f[1].z); a[7]  = f2bf(q.f[1].w);
  a[8]  = f2bf(q.f[2].x); a[9]  = f2bf(q.f[2].y); a[10] = f2bf(q.f[2].z); a[11] = f2bf(q.f[2].w);
  a[12] = f2bf(q.f[3].x); a[13] = f2bf(q.f[3].y); a[14] = f2bf(q.f[3].z); a[15] = f2bf(q.f[3].w);
  return a;
}

// ---------------------------------------------------------------------------
// Transpose + convert weights to bf16:  Wt[n][k] = (bf16) W[k][n]
// ---------------------------------------------------------------------------
__global__ __launch_bounds__(256) void prep_w_kernel(
    const float* __restrict__ W1, const float* __restrict__ W2,
    __bf16* __restrict__ Wt1, __bf16* __restrict__ Wt2) {
  int idx = blockIdx.x * 256 + threadIdx.x;
  if (idx >= NFEAT * NHID) return;
  int n = idx >> 8;
  int k = idx & 255;
  Wt1[n * NFEAT + k] = f2bf(W1[k * NHID + n]);
  Wt2[n * NFEAT + k] = f2bf(W2[k * NHID + n]);
}

// ---------------------------------------------------------------------------
// support = X (f32, [nrows x 256]) @ W (bf16 Wt[n][k]), f32 accumulate.
// Block = 8 waves covering 8 M-tiles x one 64-col N-strip. The strip's B
// panel (64x256 bf16 = 32 KB) is staged once in LDS (padded stride, conflict
// free), so the K loop reads B via ds_load_b128 while A global loads run a
// depth-1 software pipeline ahead of the 4 WMMAs per step.
// ---------------------------------------------------------------------------
__global__ __launch_bounds__(256) void gemm_bf16_kernel(
    const float* __restrict__ X, const __bf16* __restrict__ Wt,
    float* __restrict__ S, int nrows) {
  __shared__ __bf16 Bs[64 * KPAD];               // 33 KB

  const int wave = threadIdx.x >> 5;
  const int lane = threadIdx.x & 31;
  const int nt   = blockIdx.x & 3;               // N-strip 0..3
  const int mt   = (blockIdx.x >> 2) * 8 + wave; // M-tile for this wave

  // ---- cooperative LDS fill: 2048 x uint4, 8 per thread -------------------
  const __bf16* wsrc = Wt + (size_t)(nt * 64) * NFEAT;
#pragma unroll
  for (int i = 0; i < 8; ++i) {
    int idx = threadIdx.x + i * 256;             // 0..2047
    int col = idx >> 5;                          // 0..63
    int kc  = idx & 31;                          // 8-element chunk
    uint4 d = *(const uint4*)(wsrc + (size_t)col * NFEAT + kc * 8);
    *(uint4*)&Bs[col * KPAD + kc * 8] = d;
  }
  __syncthreads();

  if (mt * 16 >= nrows) return;

  const int mlane = lane & 15;                   // M (A) / N (B) within tile
  const int hi    = lane >> 4;
  const int kb    = hi * 8;                      // per-lane K sub-offset
  const int row   = mt * 16 + mlane;

  const float* xbase = X + (size_t)row * NFEAT + kb;
  const __bf16* bs0 = &Bs[( 0 + mlane) * KPAD + kb];
  const __bf16* bs1 = &Bs[(16 + mlane) * KPAD + kb];
  const __bf16* bs2 = &Bs[(32 + mlane) * KPAD + kb];
  const __bf16* bs3 = &Bs[(48 + mlane) * KPAD + kb];

  v8f acc0 = {}, acc1 = {}, acc2 = {}, acc3 = {};

  AFrag ac = load_a(xbase);
#pragma unroll
  for (int k = 0; k < NFEAT / 32; ++k) {
    AFrag an;
    if (k < NFEAT / 32 - 1) an = load_a(xbase + (k + 1) * 32);
    const int koff = k * 32;
    v16bf a = cvt_a(ac);
    BFrag b0 = load_b_lds(bs0 + koff);
    BFrag b1 = load_b_lds(bs1 + koff);
    BFrag b2 = load_b_lds(bs2 + koff);
    BFrag b3 = load_b_lds(bs3 + koff);
    acc0 = __builtin_amdgcn_wmma_f32_16x16x32_bf16(false, a, false, b0.v, (short)0, acc0, false, false);
    acc1 = __builtin_amdgcn_wmma_f32_16x16x32_bf16(false, a, false, b1.v, (short)0, acc1, false, false);
    acc2 = __builtin_amdgcn_wmma_f32_16x16x32_bf16(false, a, false, b2.v, (short)0, acc2, false, false);
    acc3 = __builtin_amdgcn_wmma_f32_16x16x32_bf16(false, a, false, b3.v, (short)0, acc3, false, false);
    ac = an;                                     // SSA rename under full unroll
  }

  // ---- store: VGPR v -> M = v + 8*hi ; N = mlane --------------------------
#pragma unroll
  for (int v = 0; v < 8; ++v) {
    const size_t r = (size_t)(mt * 16 + v + hi * 8) * NHID + nt * 64 + mlane;
    S[r +  0] = acc0[v];
    S[r + 16] = acc1[v];
    S[r + 32] = acc2[v];
    S[r + 48] = acc3[v];
  }
}

// ---------------------------------------------------------------------------
// Zero fill
// ---------------------------------------------------------------------------
__global__ __launch_bounds__(256) void fill_zero_kernel(float* __restrict__ p,
                                                        long long n) {
  long long i = (long long)blockIdx.x * 256 + threadIdx.x;
  if (i < n) p[i] = 0.0f;
}

// ---------------------------------------------------------------------------
// COO SpMM: Agg[rows[e]] += vals[e] * S[cols[e]]
// 64 threads x float4 per edge; 4 edges concurrently per 256-thread block.
// Edge-list loads are wave-uniform (one broadcast request per wave).
// ---------------------------------------------------------------------------
__global__ __launch_bounds__(256) void spmm_kernel(
    const int* __restrict__ rows, const int* __restrict__ cols,
    const float* __restrict__ vals, const float* __restrict__ S,
    float* __restrict__ Agg, int nE, int epb) {
  const int sub = threadIdx.x >> 6;              // edge slot 0..3
  const int f4  = (threadIdx.x & 63) * 4;        // feature group
  int e1 = blockIdx.x * epb + epb;
  if (e1 > nE) e1 = nE;
  for (int e = blockIdx.x * epb + sub; e < e1; e += 4) {
    const int   r = rows[e];
    const int   c = cols[e];
    const float v = vals[e];
    if (e + 4 < e1) {
      // gfx1250 global_prefetch_b8: pull next edge's feature row toward cache
      __builtin_prefetch(&S[(size_t)cols[e + 4] * NHID + f4], 0, 0);
    }
    const float4 s = *(const float4*)&S[(size_t)c * NHID + f4];
    float* dst = &Agg[(size_t)r * NHID + f4];
    atomicAdd(dst + 0, v * s.x);
    atomicAdd(dst + 1, v * s.y);
    atomicAdd(dst + 2, v * s.z);
    atomicAdd(dst + 3, v * s.w);
  }
}

// ---------------------------------------------------------------------------
// h = relu(agg + b), in place
// ---------------------------------------------------------------------------
__global__ __launch_bounds__(256) void bias_relu_kernel(
    float* __restrict__ p, const float* __restrict__ bias, long long n) {
  long long i = (long long)blockIdx.x * 256 + threadIdx.x;
  if (i >= n) return;
  float v = p[i] + bias[i & 255];
  p[i] = v > 0.0f ? v : 0.0f;
}

// ---------------------------------------------------------------------------
// out[pos[i]] = relu(agg[i] + b)   (out pre-zeroed for the padding rows)
// ---------------------------------------------------------------------------
__global__ __launch_bounds__(256) void bias_relu_scatter_kernel(
    const float* __restrict__ agg, const float* __restrict__ bias,
    const int* __restrict__ pos, float* __restrict__ out, long long n) {
  long long i = (long long)blockIdx.x * 256 + threadIdx.x;
  if (i >= n) return;
  int rowi = (int)(i >> 8);
  int f = (int)(i & 255);
  float v = agg[i] + bias[f];
  out[(size_t)pos[rowi] * NHID + f] = v > 0.0f ? v : 0.0f;
}

// ---------------------------------------------------------------------------
extern "C" void kernel_launch(void* const* d_in, const int* in_sizes, int n_in,
                              void* d_out, int out_size, void* d_ws, size_t ws_size,
                              hipStream_t stream) {
  const float* x    = (const float*)d_in[0];
  const int*   rows = (const int*)d_in[1];
  const int*   cols = (const int*)d_in[2];
  const float* vals = (const float*)d_in[3];
  const int*   pos  = (const int*)d_in[5];
  const float* W1   = (const float*)d_in[6];
  const float* b1   = (const float*)d_in[7];
  const float* W2   = (const float*)d_in[8];
  const float* b2   = (const float*)d_in[9];

  const int nrows = in_sizes[0] / NFEAT;   // 50000
  const int nE    = in_sizes[1];           // 1.6M
  const long long feat_elems = (long long)nrows * NHID;

  // Workspace layout: 2x bf16 weight panels + support/agg f32 buffers
  char* ws = (char*)d_ws;
  __bf16* Wt1  = (__bf16*)ws;                                    // 128 KB
  __bf16* Wt2  = (__bf16*)(ws + 131072);                         // 128 KB
  float*  bufA = (float*)(ws + 262144);                          // support
  float*  bufB = (float*)(ws + 262144 + (size_t)feat_elems * 4); // agg / h

  const int tilesM     = (nrows + 15) / 16;                      // 3125
  const int gemmBlocks = ((tilesM + 7) / 8) * (NHID / 64);       // 8 M-tiles/block

  const int EPB        = 256;
  const int spmmBlocks = (nE + EPB - 1) / EPB;
  const int featBlocks = (int)((feat_elems + 255) / 256);
  const int outBlocks  = (out_size + 255) / 256;

  // ---- prep weights ----
  prep_w_kernel<<<(NFEAT * NHID + 255) / 256, 256, 0, stream>>>(W1, W2, Wt1, Wt2);

  // ---- layer 1 ----
  gemm_bf16_kernel<<<gemmBlocks, 256, 0, stream>>>(x, Wt1, bufA, nrows);
  fill_zero_kernel<<<featBlocks, 256, 0, stream>>>(bufB, feat_elems);
  spmm_kernel<<<spmmBlocks, 256, 0, stream>>>(rows, cols, vals, bufA, bufB, nE, EPB);
  bias_relu_kernel<<<featBlocks, 256, 0, stream>>>(bufB, b1, feat_elems);

  // ---- layer 2 ----
  gemm_bf16_kernel<<<gemmBlocks, 256, 0, stream>>>(bufB, Wt2, bufA, nrows);
  fill_zero_kernel<<<featBlocks, 256, 0, stream>>>(bufB, feat_elems);
  spmm_kernel<<<spmmBlocks, 256, 0, stream>>>(rows, cols, vals, bufA, bufB, nE, EPB);

  // ---- scatter into padded output ----
  fill_zero_kernel<<<outBlocks, 256, 0, stream>>>((float*)d_out, out_size);
  bias_relu_scatter_kernel<<<featBlocks, 256, 0, stream>>>(bufB, b2, pos,
                                                           (float*)d_out, feat_elems);
}
